// PCENTransform_47545287967334
// MI455X (gfx1250) — matrix operand order repaired and checked by reading, not compiled, not run on hardware
//
#include <hip/hip_runtime.h>
#include <math.h>

typedef __attribute__((ext_vector_type(16))) _Float16 v16h;
typedef __attribute__((ext_vector_type(8)))  float    v8f;

namespace {
constexpr int kBatch = 128;
constexpr int kL     = 128000;
constexpr int kNFFT  = 1024;
constexpr int kHop   = 320;
constexpr int kT     = 401;          // frames per batch
constexpr int kMels  = 128;
constexpr int kFreq  = 513;          // rfft bins
constexpr int kNPad  = 528;          // GEMM1 N padded to 33 tiles of 16
constexpr int kNT1   = 33;
constexpr int kKT1   = 32;           // 1024 / 32
constexpr int kKPad2 = 544;          // GEMM2 K padded to 17 tiles of 32
constexpr int kKT2   = 17;
constexpr int kNT2   = 8;            // 128 / 16
constexpr int kRows  = kBatch * kT;  // 51328 = 3208 * 16 exactly
constexpr int kRowTiles = kRows / 16;
constexpr float kTwoPiOverN = 6.283185307179586f / 1024.0f;
}

union PackH2 { uint32_t u; _Float16 h[2]; };

// ---------------------------------------------------------------------------
// Build DFT cos / -sin matrices, pre-swizzled into WMMA B-fragment order:
//   storage dword index = ((nt*KT1 + kt)*32 + lane)*8 + v
//   element mapping:  N = nt*16 + (lane&15),
//                     K = kt*32 + ((lane>>4)<<4) + 2v + {0,1}
__global__ void k_build_dft(uint32_t* __restrict__ cosw, uint32_t* __restrict__ sinw) {
  int idx = blockIdx.x * blockDim.x + threadIdx.x;
  int total = kNT1 * kKT1 * 32 * 8;
  if (idx >= total) return;
  int v    = idx & 7;
  int lane = (idx >> 3) & 31;
  int tile = idx >> 8;
  int kt = tile % kKT1;
  int nt = tile / kKT1;
  int N  = nt * 16 + (lane & 15);
  int K  = kt * 32 + ((lane >> 4) << 4) + (v << 1);
  PackH2 pc, ps;
  pc.h[0] = pc.h[1] = ps.h[0] = ps.h[1] = (_Float16)0.0f;
  if (N < kFreq) {
    // exact phase via integer modulus: 2*pi*K*N/1024 has period 1024 in K*N
    int r0 = (K * N) & 1023;
    int r1 = ((K + 1) * N) & 1023;
    float t0 = (float)r0 * kTwoPiOverN;
    float t1 = (float)r1 * kTwoPiOverN;
    pc.h[0] = (_Float16)cosf(t0);
    pc.h[1] = (_Float16)cosf(t1);
    ps.h[0] = (_Float16)(-sinf(t0));   // Im part of rfft
    ps.h[1] = (_Float16)(-sinf(t1));
  }
  cosw[idx] = pc.u;
  sinw[idx] = ps.u;
}

// ---------------------------------------------------------------------------
// Mel filterbank (HTK scale, torchaudio defaults), same B-fragment swizzle.
__device__ __forceinline__ float mel_fpt(int j) {
  float melmax = 2595.0f * log10f(1.0f + 16000.0f / 700.0f);
  float m = melmax * (float)j * (1.0f / 129.0f);
  return 700.0f * (powf(10.0f, m * (1.0f / 2595.0f)) - 1.0f);
}

__global__ void k_build_fb(uint32_t* __restrict__ fbw) {
  int idx = blockIdx.x * blockDim.x + threadIdx.x;
  int total = kNT2 * kKT2 * 32 * 8;
  if (idx >= total) return;
  int v    = idx & 7;
  int lane = (idx >> 3) & 31;
  int tile = idx >> 8;
  int kt = tile % kKT2;
  int nt = tile / kKT2;
  int m  = nt * 16 + (lane & 15);                     // mel index, < 128
  int K  = kt * 32 + ((lane >> 4) << 4) + (v << 1);   // freq bin, < 544
  float p0 = mel_fpt(m), p1 = mel_fpt(m + 1), p2 = mel_fpt(m + 2);
  PackH2 p;
  #pragma unroll
  for (int q = 0; q < 2; ++q) {
    int Kq = K + q;
    float val = 0.0f;
    if (Kq < kFreq) {
      float f    = (float)Kq * 31.25f;                // SR/2 / 512
      float down = (f - p0) / (p1 - p0);
      float up   = (p2 - f) / (p2 - p1);
      val = fmaxf(0.0f, fminf(down, up));
    }
    p.h[q] = (_Float16)val;
  }
  fbw[idx] = p.u;
}

// ---------------------------------------------------------------------------
// A-fragment gather from row-major LDS tile, contiguous k order, same as the B packing:
//   M = lane&15;  lanes 0-15 hold K {0..15}, lanes 16-31 {16..31}
__device__ __forceinline__ v16h load_a_frag(const _Float16* lds, int ldk, int k0, int lane) {
  int m  = lane & 15;
  int kb = (lane >> 4) << 4;
  v16h a;
  uint32_t* ap = (uint32_t*)&a;
  #pragma unroll
  for (int v = 0; v < 8; ++v) {
    int K = k0 + kb + (v << 1);
    ap[v] = *(const uint32_t*)(&lds[m * ldk + K]);    // two f16 per dword
  }
  return a;
}

// ---------------------------------------------------------------------------
// Fused: frame+window -> |DFT| via WMMA GEMM -> mel via WMMA GEMM.
// One wave32 per 16-frame tile.
__global__ void __launch_bounds__(32)
k_melgemm(const float* __restrict__ x,
          const v16h* __restrict__ cosw,
          const v16h* __restrict__ sinw,
          const v16h* __restrict__ fbw,
          float* __restrict__ mel) {
  __shared__ _Float16 fr[16 * kNFFT];    // 32 KB  windowed frames
  __shared__ _Float16 mg[16 * kKPad2];   // 17 KB  magnitude spectrum (f16)
  const int lane = threadIdx.x;
  const int tile = blockIdx.x;

  // Stage frames with reflect-pad + periodic Hann window.
  for (int i = lane; i < 16 * kNFFT; i += 32) {
    int row = i >> 10;
    int k   = i & 1023;
    int g = tile * 16 + row;
    int b = g / kT;
    int t = g - b * kT;
    int p = t * kHop + k - 512;            // center=True: pad = n_fft/2
    if (p < 0) p = -p;
    else if (p >= kL) p = 2 * kL - 2 - p;
    float w = 0.5f - 0.5f * cosf((float)k * (3.14159265358979f / 512.0f));
    fr[i] = (_Float16)(x[b * kL + p] * w);
  }
  for (int i = lane; i < 16 * kKPad2; i += 32) mg[i] = (_Float16)0.0f;
  __syncthreads();

  // GEMM1: Re/Im accumulation, then magnitude into LDS.
  for (int nt = 0; nt < kNT1; ++nt) {
    v8f accR = {};
    v8f accI = {};
    #pragma unroll 2
    for (int kt = 0; kt < kKT1; ++kt) {
      v16h a  = load_a_frag(fr, kNFFT, kt * 32, lane);
      v16h bc = cosw[(nt * kKT1 + kt) * 32 + lane];
      v16h bs = sinw[(nt * kKT1 + kt) * 32 + lane];
      accR = __builtin_amdgcn_wmma_f32_16x16x32_f16(false, a, false, bc, (short)0, accR, false, false);
      accI = __builtin_amdgcn_wmma_f32_16x16x32_f16(false, a, false, bs, (short)0, accI, false, false);
    }
    int row0 = (lane >> 4) << 3;
    int col  = lane & 15;
    #pragma unroll
    for (int r = 0; r < 8; ++r) {
      float magv = sqrtf(accR[r] * accR[r] + accI[r] * accI[r]);
      mg[(row0 + r) * kKPad2 + nt * 16 + col] = (_Float16)magv;
    }
  }
  __syncthreads();

  // GEMM2: mel projection, scatter to [b][mel][t].
  for (int nt = 0; nt < kNT2; ++nt) {
    v8f acc = {};
    #pragma unroll 2
    for (int kt = 0; kt < kKT2; ++kt) {
      v16h a  = load_a_frag(mg, kKPad2, kt * 32, lane);
      v16h bf = fbw[(nt * kKT2 + kt) * 32 + lane];
      acc = __builtin_amdgcn_wmma_f32_16x16x32_f16(false, a, false, bf, (short)0, acc, false, false);
    }
    int row0 = (lane >> 4) << 3;
    int col  = lane & 15;
    #pragma unroll
    for (int r = 0; r < 8; ++r) {
      int g = tile * 16 + row0 + r;
      int b = g / kT;
      int t = g - b * kT;
      mel[((size_t)b * kMels + nt * 16 + col) * kT + t] = acc[r];
    }
  }
}

// ---------------------------------------------------------------------------
// PCEN: sequential IIR per (b, mel) row; deterministic per-row partial sums.
__global__ void k_pcen(const float* __restrict__ mel, float* __restrict__ out,
                       float* __restrict__ rowsum, float* __restrict__ rowsq) {
  int row = blockIdx.x * blockDim.x + threadIdx.x;
  if (row >= kBatch * kMels) return;
  const float* mrow = mel + (size_t)row * kT;
  float* orow = out + (size_t)row * kT;
  float M = 0.0f, s = 0.0f, s2 = 0.0f;
  const float sq2 = 1.4142135623730951f;          // delta^r, delta=2, r=0.5
  for (int t = 0; t < kT; ++t) {
    float xv = mrow[t];
    M = (t == 0) ? xv : 0.975f * M + 0.025f * xv; // s = 0.025
    float smooth = powf(1e-6f + M, 0.98f);        // alpha = 0.98
    float pc = sqrtf(xv / smooth + 2.0f) - sq2;
    orow[t] = pc;
    s  += pc;
    s2 += pc * pc;
  }
  rowsum[row] = s;
  rowsq[row]  = s2;
}

// Per-batch mean / 1/(std+eps), ddof=1, deterministic LDS tree reduction.
__global__ void k_stats(const float* __restrict__ rowsum, const float* __restrict__ rowsq,
                        float* __restrict__ bmean, float* __restrict__ binv) {
  __shared__ float s1[128];
  __shared__ float s2[128];
  int b = blockIdx.x, t = threadIdx.x;
  s1[t] = rowsum[b * kMels + t];
  s2[t] = rowsq[b * kMels + t];
  __syncthreads();
  for (int o = 64; o > 0; o >>= 1) {
    if (t < o) { s1[t] += s1[t + o]; s2[t] += s2[t + o]; }
    __syncthreads();
  }
  if (t == 0) {
    float n = (float)(kMels * kT);                 // 51328
    float mean = s1[0] / n;
    float var  = (s2[0] - s1[0] * s1[0] / n) / (n - 1.0f);
    var = fmaxf(var, 0.0f);
    bmean[b] = mean;
    binv[b]  = 1.0f / (sqrtf(var) + 1e-6f);
  }
}

__global__ void k_norm(float* __restrict__ out,
                       const float* __restrict__ bmean, const float* __restrict__ binv) {
  int i = blockIdx.x * blockDim.x + threadIdx.x;
  if (i >= kBatch * kMels * kT) return;
  int b = i / (kMels * kT);
  out[i] = (out[i] - bmean[b]) * binv[b];
}

// ---------------------------------------------------------------------------
extern "C" void kernel_launch(void* const* d_in, const int* in_sizes, int n_in,
                              void* d_out, int out_size, void* d_ws, size_t ws_size,
                              hipStream_t stream) {
  (void)in_sizes; (void)n_in; (void)out_size;
  const float* x = (const float*)d_in[0];
  float* out = (float*)d_out;
  char* ws = (char*)d_ws;

  auto al = [](size_t v) { return (v + 255) & ~(size_t)255; };
  constexpr size_t COS_BYTES = (size_t)kNT1 * kKT1 * 256 * 4;  // 1,081,344
  constexpr size_t FB_BYTES  = (size_t)kNT2 * kKT2 * 256 * 4;  //   139,264
  constexpr size_t MEL_BYTES = (size_t)kRows * kMels * 4;      // 26,279,936
  size_t cos_off = 0;
  size_t sin_off = al(cos_off + COS_BYTES);
  size_t fb_off  = al(sin_off + COS_BYTES);
  size_t mel_off = al(fb_off + FB_BYTES);
  size_t rs_off  = al(mel_off + MEL_BYTES);
  size_t rq_off  = al(rs_off + (size_t)kBatch * kMels * 4);
  size_t bm_off  = al(rq_off + (size_t)kBatch * kMels * 4);
  size_t bi_off  = al(bm_off + (size_t)kBatch * 4);
  (void)ws_size;  // ~28.7 MB total required

  uint32_t* cosw = (uint32_t*)(ws + cos_off);
  uint32_t* sinw = (uint32_t*)(ws + sin_off);
  uint32_t* fbw  = (uint32_t*)(ws + fb_off);
  float* mel     = (float*)(ws + mel_off);
  float* rowsum  = (float*)(ws + rs_off);
  float* rowsq   = (float*)(ws + rq_off);
  float* bmean   = (float*)(ws + bm_off);
  float* binv    = (float*)(ws + bi_off);

  {
    int total = kNT1 * kKT1 * 256;
    k_build_dft<<<(total + 255) / 256, 256, 0, stream>>>(cosw, sinw);
  }
  {
    int total = kNT2 * kKT2 * 256;
    k_build_fb<<<(total + 255) / 256, 256, 0, stream>>>(fbw);
  }
  k_melgemm<<<kRowTiles, 32, 0, stream>>>(x, (const v16h*)cosw, (const v16h*)sinw,
                                          (const v16h*)fbw, mel);
  k_pcen<<<(kBatch * kMels + 255) / 256, 256, 0, stream>>>(mel, out, rowsum, rowsq);
  k_stats<<<kBatch, 128, 0, stream>>>(rowsum, rowsq, bmean, binv);
  {
    int total = kBatch * kMels * kT;
    k_norm<<<(total + 255) / 256, 256, 0, stream>>>(out, bmean, binv);
  }
}
